// GTUBlock2D_37795712204995
// MI455X (gfx1250) — compile-verified, matrix operand order checked by reading, not run on hardware
//
#include <hip/hip_runtime.h>
#include <hip/hip_bf16.h>
#include <math.h>

// ---------------- problem constants ----------------
constexpr int E   = 512;
constexpr int D1  = 1536;
constexpr int H   = 8;
constexpr int DH  = 192;   // D1 / H
constexpr int KNN = 16;
constexpr int RPE = 32;
constexpr int B   = 4;
constexpr int N   = 1024;
constexpr int NPOS = 2 * KNN - 1;  // 31 toeplitz offsets

typedef __bf16 bf16;
typedef __attribute__((ext_vector_type(16))) __bf16 v16bf;
typedef __attribute__((ext_vector_type(8)))  float  v8f;

// Build a 16-element bf16 fragment from two contiguous 8-element (16B) runs.
// Matches the CDNA5 16-bit operand layout: per lane, K = half*8+0..7 and
// K = 16+half*8+0..7 within a 32-wide K block.
__device__ __forceinline__ v16bf ld_frag(const bf16* p0, const bf16* p1) {
    v16bf r;
#pragma unroll
    for (int i = 0; i < 8; ++i) { r[i] = p0[i]; r[i + 8] = p1[i]; }
    return r;
}

// Fast SiLU: x * sigmoid(x) using v_exp_f32 + v_rcp_f32 (avoids the IEEE
// v_div_scale/v_div_fmas sequence in the per-tile epilogue).
__device__ __forceinline__ float silu_f(float x) {
    return x * __builtin_amdgcn_rcpf(1.0f + __expf(-x));
}

// ---------------- kernel 0: weight convert/transpose to bf16 ----------------
// wuT[n*E+k] = wu[k*D1+n]; wvT same; woT[n*D1+k] = wo[k*E+n]
__global__ void gtu_convert(const float* __restrict__ wu,
                            const float* __restrict__ wv,
                            const float* __restrict__ wo,
                            bf16* __restrict__ wuT,
                            bf16* __restrict__ wvT,
                            bf16* __restrict__ woT) {
    int idx = blockIdx.x * blockDim.x + threadIdx.x;
    if (idx >= E * D1) return;
    {
        int n = idx / E, k = idx - n * E;            // n < D1, k < E
        wuT[idx] = (bf16)wu[(size_t)k * D1 + n];
        wvT[idx] = (bf16)wv[(size_t)k * D1 + n];
    }
    {
        int n = idx / D1, k = idx - n * D1;          // n < E, k < D1
        woT[idx] = (bf16)wo[(size_t)k * E + n];
    }
}

// ---------------- kernel 1: RPE MLP -> net[31][D1] ----------------
__global__ void gtu_rpe(const float* __restrict__ wp, const float* __restrict__ bp,
                        const float* __restrict__ ws, const float* __restrict__ bs,
                        const float* __restrict__ wo2, const float* __restrict__ bo2,
                        float* __restrict__ net) {
    __shared__ float hbuf[RPE];
    __shared__ float gbuf[RPE];
    const int tid = threadIdx.x;
    const float posv = (float)blockIdx.x - (float)(KNN - 1);  // offsets -15..15

    if (tid < RPE) hbuf[tid] = posv * wp[tid] + bp[tid];
    __syncthreads();

    for (int l = 0; l < 3; ++l) {
        float yv = 0.0f;
        if (tid < RPE) {
            float ssq = 0.0f;
            for (int i = 0; i < RPE; ++i) ssq += hbuf[i] * hbuf[i];
            float rms = sqrtf(ssq) * 0.17677669529663687f;    // * RPE^-0.5
            float scale = 1.0f / (rms + 1e-8f);
            float acc = bs[l * RPE + tid];
            for (int i = 0; i < RPE; ++i) {
                float a = hbuf[i] * scale;
                a = a > 0.0f ? a : 0.0f;
                acc += a * ws[(l * RPE + i) * RPE + tid];
            }
            yv = acc;
        }
        __syncthreads();
        if (tid < RPE) hbuf[tid] = yv;
        __syncthreads();
    }
    if (tid < RPE) {
        float ssq = 0.0f;
        for (int i = 0; i < RPE; ++i) ssq += hbuf[i] * hbuf[i];
        float rms = sqrtf(ssq) * 0.17677669529663687f;
        float scale = 1.0f / (rms + 1e-8f);
        float a = hbuf[tid] * scale;
        gbuf[tid] = a > 0.0f ? a : 0.0f;
    }
    __syncthreads();
    for (int o = tid; o < D1; o += blockDim.x) {
        float acc = bo2[o];
        for (int i = 0; i < RPE; ++i) acc += gbuf[i] * wo2[(size_t)i * D1 + o];
        net[(size_t)blockIdx.x * D1 + o] = acc;
    }
}

// ---------------- kernel 2: KNN (top-16 smallest sq-dist) ----------------
__global__ void gtu_knn(const float* __restrict__ xyz, int* __restrict__ knn) {
    __shared__ float px[256], py[256], pz[256];
    const int tid = threadIdx.x;
    const int q = blockIdx.x * 128 + tid;       // flattened (b,n)
    const int b = q >> 10;
    const float* xyzb = xyz + (size_t)(b << 10) * 3;

    const float qx = xyz[(size_t)q * 3 + 0];
    const float qy = xyz[(size_t)q * 3 + 1];
    const float qz = xyz[(size_t)q * 3 + 2];

    float bd[KNN]; int bi[KNN];
#pragma unroll
    for (int j = 0; j < KNN; ++j) { bd[j] = 3.4e38f; bi[j] = 0; }

    for (int m0 = 0; m0 < N; m0 += 256) {
        for (int l = tid; l < 256; l += 128) {
            px[l] = xyzb[(size_t)(m0 + l) * 3 + 0];
            py[l] = xyzb[(size_t)(m0 + l) * 3 + 1];
            pz[l] = xyzb[(size_t)(m0 + l) * 3 + 2];
        }
        __syncthreads();
        for (int mm = 0; mm < 256; ++mm) {
            float dx = qx - px[mm], dy = qy - py[mm], dz = qz - pz[mm];
            float d = dx * dx + dy * dy + dz * dz;
            if (d < bd[KNN - 1]) {
                bd[KNN - 1] = d; bi[KNN - 1] = m0 + mm;
#pragma unroll
                for (int j = KNN - 1; j > 0; --j) {
                    if (bd[j] < bd[j - 1]) {
                        float td = bd[j]; bd[j] = bd[j - 1]; bd[j - 1] = td;
                        int   ti = bi[j]; bi[j] = bi[j - 1]; bi[j - 1] = ti;
                    }
                }
            }
        }
        __syncthreads();
    }
#pragma unroll
    for (int j = 0; j < KNN; ++j) knn[(size_t)q * KNN + j] = bi[j];
}

// ---------------- kernel 3: gather+norm+GLU GEMMs (WMMA) + toeplitz mix ----
// One block per token (b,n); 256 threads = 8 waves; wave owns 12 D1-tiles.
__global__ void __launch_bounds__(256)
gtu_main(const float* __restrict__ x,
         const int* __restrict__ knn,
         const bf16* __restrict__ wuT,
         const bf16* __restrict__ wvT,
         const float* __restrict__ bu,
         const float* __restrict__ bv,
         const float* __restrict__ net,
         float* __restrict__ red) {
    __shared__ __align__(16) bf16 As[KNN][E];   // 16 KB normalized A-panel
    __shared__ float psum[256];
    __shared__ float rscale[KNN];

    const int tid  = threadIdx.x;
    const int wave = tid >> 5;
    const int lane = tid & 31;
    const int half = lane >> 4;
    const int col  = lane & 15;
    const int bn   = blockIdx.x;
    const int b    = bn >> 10;

    // ---- stage A: gather neighbor rows, compute srms(., D1) scale ----
    const int r     = tid & 15;        // neighbor row
    const int chunk = tid >> 4;        // 16 chunks of 32 columns
    const int g     = knn[(size_t)bn * KNN + r];
    const float* xr = x + ((size_t)(b << 10) + g) * E;
    const int c0 = chunk * 32;
    float s = 0.0f;
#pragma unroll 8
    for (int c = 0; c < 32; ++c) { float v = xr[c0 + c]; s += v * v; }
    psum[tid] = s;
    __syncthreads();
    if (tid < KNN) {
        float t = 0.0f;
#pragma unroll
        for (int j = 0; j < 16; ++j) t += psum[j * 16 + tid];
        float rms = sqrtf(t) * 0.02551551815399144f;   // * D1^-0.5 (mismatched dim, per reference)
        rscale[tid] = __builtin_amdgcn_rcpf(rms + 1e-8f);
    }
    __syncthreads();
    {
        float sc = rscale[r];
#pragma unroll 8
        for (int c = 0; c < 32; ++c) As[r][c0 + c] = (bf16)(xr[c0 + c] * sc);
    }
    __syncthreads();

    // ---- per-wave: 12 tiles of 16 channels; u,v WMMA then fused toeplitz ----
    const bf16* arow = &As[col][0];                    // A row = lane%16
    for (int tt = 0; tt < 12; ++tt) {
        const int dt = wave * 12 + tt;
        const int dg = dt * 16 + col;                  // global channel
        v8f accu, accv;
        {
            float bub = bu[dg], bvb = bv[dg];
#pragma unroll
            for (int r8 = 0; r8 < 8; ++r8) { accu[r8] = bub; accv[r8] = bvb; }
        }
        const bf16* bu_base = wuT + (size_t)dg * E;
        const bf16* bv_base = wvT + (size_t)dg * E;
#pragma unroll 4
        for (int kb = 0; kb < E / 32; ++kb) {
            const int k0 = kb * 32 + half * 8;
            v16bf af  = ld_frag(arow + k0,    arow + k0 + 16);
            v16bf bfu = ld_frag(bu_base + k0, bu_base + k0 + 16);
            v16bf bfv = ld_frag(bv_base + k0, bv_base + k0 + 16);
            accu = __builtin_amdgcn_wmma_f32_16x16x32_bf16(
                false, af, false, bfu, (short)0, accu, false, false);
            accv = __builtin_amdgcn_wmma_f32_16x16x32_bf16(
                false, af, false, bfv, (short)0, accv, false, false);
        }
        // SiLU (fast: v_exp + v_rcp)
#pragma unroll
        for (int r8 = 0; r8 < 8; ++r8) { accu[r8] = silu_f(accu[r8]); accv[r8] = silu_f(accv[r8]); }

        // gather full v column (16 neighbor rows) via cross-half shuffle
        float vlo[8], vhi[8];
#pragma unroll
        for (int r8 = 0; r8 < 8; ++r8) {
            float own = accv[r8];
            float oth = __shfl_xor(own, 16, 32);
            vlo[r8] = half ? oth : own;   // rows 0..7
            vhi[r8] = half ? own : oth;   // rows 8..15
        }
        // hoist the 23 toeplitz coefficients this lane needs
        float nv[23];
#pragma unroll
        for (int t2 = 0; t2 < 23; ++t2)
            nv[t2] = net[(size_t)(half * 8 + t2) * D1 + dg];

        // red[dg] = sum_i u[i] * sum_j a_{i-j} * v[j]   (i restricted to this half)
        float part = 0.0f;
#pragma unroll
        for (int ii = 0; ii < 8; ++ii) {
            float ssum = 0.0f;
#pragma unroll
            for (int j = 0; j < 16; ++j) {
                float vjv = (j < 8) ? vlo[j] : vhi[j - 8];
                ssum += nv[ii - j + 15] * vjv;
            }
            part += accu[ii] * ssum;
        }
        float tot = part + __shfl_xor(part, 16, 32);
        if (half == 0) red[(size_t)bn * D1 + dg] = tot;
    }
}

// ---------------- kernel 4: out = red @ wo + bo + shortcut (WMMA) ----------
__global__ void __launch_bounds__(256)
gtu_proj(const float* __restrict__ red,
         const bf16* __restrict__ woT,
         const float* __restrict__ bo,
         const float* __restrict__ x,
         float* __restrict__ out) {
    __shared__ __align__(16) bf16 Ar[16][D1];   // 48 KB
    const int tid  = threadIdx.x;
    const int wave = tid >> 5;
    const int lane = tid & 31;
    const int half = lane >> 4;
    const int col  = lane & 15;
    const int t0   = blockIdx.x * 16;           // 16 tokens per block

    for (int idx = tid; idx < 16 * D1; idx += 256) {
        int rr = idx / D1, cc = idx - rr * D1;
        Ar[rr][cc] = (bf16)red[(size_t)(t0 + rr) * D1 + cc];
    }
    __syncthreads();

    const bf16* arow = &Ar[col][0];
    for (int q = 0; q < 4; ++q) {
        const int nt = wave * 4 + q;
        const int ncol = nt * 16 + col;
        v8f acc;
        {
            float bb = bo[ncol];
#pragma unroll
            for (int r8 = 0; r8 < 8; ++r8) acc[r8] = bb;
        }
        const bf16* bbase = woT + (size_t)ncol * D1;
#pragma unroll 4
        for (int kb = 0; kb < D1 / 32; ++kb) {
            const int k0 = kb * 32 + half * 8;
            v16bf af = ld_frag(arow + k0,  arow + k0 + 16);
            v16bf bf = ld_frag(bbase + k0, bbase + k0 + 16);
            acc = __builtin_amdgcn_wmma_f32_16x16x32_bf16(
                false, af, false, bf, (short)0, acc, false, false);
        }
#pragma unroll
        for (int r8 = 0; r8 < 8; ++r8) {
            int row = r8 + half * 8;
            size_t tok = (size_t)(t0 + row);
            out[tok * E + ncol] = acc[r8] + x[tok * E + ncol];
        }
    }
}

// ---------------- launcher ----------------
extern "C" void kernel_launch(void* const* d_in, const int* in_sizes, int n_in,
                              void* d_out, int out_size, void* d_ws, size_t ws_size,
                              hipStream_t stream) {
    (void)in_sizes; (void)n_in; (void)out_size; (void)ws_size;

    const float* xyz = (const float*)d_in[0];
    const float* x   = (const float*)d_in[1];
    const float* wu  = (const float*)d_in[2];
    const float* bu  = (const float*)d_in[3];
    const float* wv  = (const float*)d_in[4];
    const float* bv  = (const float*)d_in[5];
    const float* wo  = (const float*)d_in[6];
    const float* bo  = (const float*)d_in[7];
    const float* wp  = (const float*)d_in[8];
    const float* bp  = (const float*)d_in[9];
    const float* ws  = (const float*)d_in[10];
    const float* bs  = (const float*)d_in[11];
    const float* wo2 = (const float*)d_in[12];
    const float* bo2 = (const float*)d_in[13];

    float* out = (float*)d_out;                       // (B,N,E)
    float* red = out + (size_t)B * N * E;             // (B,N,D1) second tuple output

    // workspace layout (all 256B-aligned sizes); total ~5.1 MB
    char* w = (char*)d_ws;
    bf16* wuT = (bf16*)w;  w += (size_t)E * D1 * sizeof(bf16);   // 1.5 MB
    bf16* wvT = (bf16*)w;  w += (size_t)E * D1 * sizeof(bf16);   // 1.5 MB
    bf16* woT = (bf16*)w;  w += (size_t)E * D1 * sizeof(bf16);   // 1.5 MB
    float* net = (float*)w; w += (size_t)NPOS * D1 * sizeof(float); // 190 KB
    int* knn = (int*)w;                                           // 256 KB

    gtu_convert<<<(E * D1 + 255) / 256, 256, 0, stream>>>(wu, wv, wo, wuT, wvT, woT);
    gtu_rpe<<<NPOS, 256, 0, stream>>>(wp, bp, ws, bs, wo2, bo2, net);
    gtu_knn<<<(B * N) / 128, 128, 0, stream>>>(xyz, knn);
    gtu_main<<<B * N, 256, 0, stream>>>(x, knn, wuT, wvT, bu, bv, net, red);
    gtu_proj<<<(B * N) / 16, 256, 0, stream>>>(red, woT, bo, x, out);
}
